// IRNet_24678882083159
// MI455X (gfx1250) — matrix operand order repaired and checked by reading, not compiled, not run on hardware
//
#include <hip/hip_runtime.h>
#include <hip/hip_bf16.h>

// ---------------------------------------------------------------------------
// Types for CDNA5 WMMA (wave32)
// ---------------------------------------------------------------------------
typedef __attribute__((ext_vector_type(16))) __bf16 v16bf;
typedef __attribute__((ext_vector_type(8)))  __bf16 v8bf;
typedef __attribute__((ext_vector_type(8)))  float  v8f;

#define D_MODEL 256
#define N_HEAD  8
#define DK_HEAD 32
#define D_FF    1024
#define N_LAYER 8

// ---------------------------------------------------------------------------
// Weight convert + transpose: W[K x Nc] f32 (row-major)  ->  Wt[Nc x K] bf16
// (so the WMMA B-fragment per output column is one contiguous 32B chunk)
// ---------------------------------------------------------------------------
__global__ void transpose_to_bf16_kernel(const float* __restrict__ W,
                                         __bf16* __restrict__ Wt,
                                         int K, int Nc) {
    size_t i = (size_t)blockIdx.x * blockDim.x + threadIdx.x;
    size_t total = (size_t)K * Nc;
    if (i < total) {
        int n = (int)(i / K);
        int k = (int)(i % K);
        Wt[i] = (__bf16)W[(size_t)k * Nc + n];
    }
}

__global__ void f32_to_bf16_kernel(const float* __restrict__ in,
                                   __bf16* __restrict__ out, size_t n) {
    size_t i = (size_t)blockIdx.x * blockDim.x + threadIdx.x;
    if (i < n) out[i] = (__bf16)in[i];
}

__global__ void zero_f32_kernel(float* __restrict__ p, size_t n) {
    size_t i = (size_t)blockIdx.x * blockDim.x + threadIdx.x;
    if (i < n) p[i] = 0.0f;
}

// ---------------------------------------------------------------------------
// WMMA GEMM:  out[M x Nc] = A[M x K](bf16) * Bt[Nc x K]^T(bf16) (+bias)(relu?)
// Block: 256 threads = 8 waves.  Wave w -> rows [blk*128 + w*16, +16),
// 4 tiles of 16 cols each -> 64-col strip.  K stepped by 32.
// A fragment (16x32 bf16): lane row = lane&15,
//   elems 0..15 = K kb..kb+15, kb=(lane>>4)*16 (same k order as B).
// B fragment (32x16 bf16): lane col = lane&15, elems 0..15 = K kb..kb+15,
//   kb=(lane>>4)*16  -> contiguous in the transposed weight layout.
// C/D (16x16 f32): vgpr r -> row r + (lane>>4)*8, col = lane&15.
// ---------------------------------------------------------------------------
__global__ __launch_bounds__(256)
void gemm_bf16_wmma_kernel(const __bf16* __restrict__ A,
                           const __bf16* __restrict__ Bt,
                           const float* __restrict__ bias,
                           float* __restrict__ outF,
                           __bf16* __restrict__ outB,
                           int M, int K, int Nc, int relu) {
    const int lane = threadIdx.x & 31;
    const int wave = threadIdx.x >> 5;
    const int m0   = blockIdx.x * 128 + wave * 16;
    const int n0   = blockIdx.y * 64;

    const int rowA = m0 + (lane & 15);
    const int rA   = rowA < M ? rowA : (M - 1);   // clamp: EXEC stays all-1s
    const int kbA  = (lane >> 4) * 16;
    const int kbB  = (lane >> 4) * 16;
    const int nB   = lane & 15;

    const __bf16* __restrict__ Arow = A + (size_t)rA * K;

    v8f acc0 = {}, acc1 = {}, acc2 = {}, acc3 = {};

    for (int k0 = 0; k0 < K; k0 += 32) {
        v8bf alo = *(const v8bf*)(Arow + k0 + kbA);
        v8bf ahi = *(const v8bf*)(Arow + k0 + kbA + 8);
        v16bf af;
#pragma unroll
        for (int i = 0; i < 8; ++i) { af[i] = alo[i]; af[i + 8] = ahi[i]; }

        const __bf16* __restrict__ B0 = Bt + (size_t)(n0 +  0 + nB) * K + k0 + kbB;
        const __bf16* __restrict__ B1 = Bt + (size_t)(n0 + 16 + nB) * K + k0 + kbB;
        const __bf16* __restrict__ B2 = Bt + (size_t)(n0 + 32 + nB) * K + k0 + kbB;
        const __bf16* __restrict__ B3 = Bt + (size_t)(n0 + 48 + nB) * K + k0 + kbB;
        v16bf b0 = *(const v16bf*)B0;
        v16bf b1 = *(const v16bf*)B1;
        v16bf b2 = *(const v16bf*)B2;
        v16bf b3 = *(const v16bf*)B3;

        acc0 = __builtin_amdgcn_wmma_f32_16x16x32_bf16(false, af, false, b0, (short)0, acc0, false, false);
        acc1 = __builtin_amdgcn_wmma_f32_16x16x32_bf16(false, af, false, b1, (short)0, acc1, false, false);
        acc2 = __builtin_amdgcn_wmma_f32_16x16x32_bf16(false, af, false, b2, (short)0, acc2, false, false);
        acc3 = __builtin_amdgcn_wmma_f32_16x16x32_bf16(false, af, false, b3, (short)0, acc3, false, false);
    }

    v8f accs[4] = {acc0, acc1, acc2, acc3};
#pragma unroll
    for (int t = 0; t < 4; ++t) {
        const int n  = n0 + t * 16 + nB;
        const float bv = bias ? bias[n] : 0.0f;
#pragma unroll
        for (int r = 0; r < 8; ++r) {
            const int m = m0 + r + ((lane >> 4) << 3);
            if (m < M) {
                float v = accs[t][r] + bv;
                if (relu) v = fmaxf(v, 0.0f);
                if (outF) outF[(size_t)m * Nc + n] = v;
                if (outB) outB[(size_t)m * Nc + n] = (__bf16)v;
            }
        }
    }
}

// ---------------------------------------------------------------------------
// Edge score: per edge e, per head h:  exp(clip(<k[src],q[dst]>_h / sqrt(32)))
// wave per edge, lane = element of the 32-dim head dot product.
// ---------------------------------------------------------------------------
__global__ __launch_bounds__(256)
void edge_score_kernel(const int* __restrict__ src, const int* __restrict__ dst,
                       const float* __restrict__ q, const float* __restrict__ k,
                       float* __restrict__ score, int E) {
    const int e = blockIdx.x * 8 + (threadIdx.x >> 5);
    if (e >= E) return;
    const int lane = threadIdx.x & 31;
    const int s = src[e], d = dst[e];
    const float* __restrict__ kr = k + (size_t)s * D_MODEL;
    const float* __restrict__ qr = q + (size_t)d * D_MODEL;
#pragma unroll
    for (int h = 0; h < N_HEAD; ++h) {
        float p = kr[h * DK_HEAD + lane] * qr[h * DK_HEAD + lane];
#pragma unroll
        for (int off = 16; off; off >>= 1) p += __shfl_xor(p, off, 32);
        if (lane == 0) {
            float sc = p * 0.17677669529663687f;          // 1/sqrt(32)
            sc = fminf(fmaxf(sc, -5.0f), 5.0f);
            score[(size_t)e * N_HEAD + h] = __expf(sc);
        }
    }
}

// ---------------------------------------------------------------------------
// Edge aggregation: wv[dst] += v[src]*score ; z[dst] += score  (f32 atomics)
// ---------------------------------------------------------------------------
__global__ __launch_bounds__(256)
void edge_aggr_kernel(const int* __restrict__ src, const int* __restrict__ dst,
                      const float* __restrict__ v, const float* __restrict__ score,
                      float* __restrict__ wv, float* __restrict__ z, int E) {
    const int e = blockIdx.x * 8 + (threadIdx.x >> 5);
    if (e >= E) return;
    const int lane = threadIdx.x & 31;
    const int s = src[e], d = dst[e];
    const float* __restrict__ vr = v + (size_t)s * D_MODEL;
    float* __restrict__ wvr = wv + (size_t)d * D_MODEL;
#pragma unroll
    for (int h = 0; h < N_HEAD; ++h) {
        const float sc = score[(size_t)e * N_HEAD + h];
        unsafeAtomicAdd(&wvr[h * DK_HEAD + lane], vr[h * DK_HEAD + lane] * sc);
    }
    if (lane < N_HEAD)
        unsafeAtomicAdd(&z[(size_t)d * N_HEAD + lane], score[(size_t)e * N_HEAD + lane]);
}

// ---------------------------------------------------------------------------
// o = wv / z  -> bf16 (input to the Wo GEMM).  Wave per node.
// ---------------------------------------------------------------------------
__global__ __launch_bounds__(256)
void finalize_o_kernel(const float* __restrict__ wv, const float* __restrict__ z,
                       __bf16* __restrict__ ob, int NN) {
    const int n = blockIdx.x * 8 + (threadIdx.x >> 5);
    if (n >= NN) return;
    const int lane = threadIdx.x & 31;
#pragma unroll
    for (int h = 0; h < N_HEAD; ++h) {
        const float zz = z[(size_t)n * N_HEAD + h];
        ob[(size_t)n * D_MODEL + h * DK_HEAD + lane] =
            (__bf16)(wv[(size_t)n * D_MODEL + h * DK_HEAD + lane] / zz);
    }
}

// ---------------------------------------------------------------------------
// out = LayerNorm(resid + y) * g + b ; emits f32 (residual stream) + bf16
// (next GEMM input).  Wave per node, 8 elements per lane, shuffle reduce.
// ---------------------------------------------------------------------------
__global__ __launch_bounds__(256)
void residual_ln_kernel(const float* __restrict__ resid, const float* __restrict__ y,
                        const float* __restrict__ g, const float* __restrict__ b,
                        float* __restrict__ outF, __bf16* __restrict__ outB, int NN) {
    const int n = blockIdx.x * 8 + (threadIdx.x >> 5);
    if (n >= NN) return;
    const int lane = threadIdx.x & 31;
    float t[8];
    float s = 0.0f, ss = 0.0f;
#pragma unroll
    for (int i = 0; i < 8; ++i) {
        const int dd = i * 32 + lane;
        const float val = resid[(size_t)n * D_MODEL + dd] + y[(size_t)n * D_MODEL + dd];
        t[i] = val; s += val; ss += val * val;
    }
#pragma unroll
    for (int off = 16; off; off >>= 1) {
        s  += __shfl_xor(s,  off, 32);
        ss += __shfl_xor(ss, off, 32);
    }
    const float mean = s * (1.0f / 256.0f);
    const float var  = ss * (1.0f / 256.0f) - mean * mean;
    const float rstd = rsqrtf(var + 1e-5f);
#pragma unroll
    for (int i = 0; i < 8; ++i) {
        const int dd = i * 32 + lane;
        const float o = (t[i] - mean) * rstd * g[dd] + b[dd];
        outF[(size_t)n * D_MODEL + dd] = o;
        if (outB) outB[(size_t)n * D_MODEL + dd] = (__bf16)o;
    }
}

// ---------------------------------------------------------------------------
// Host orchestration
// ---------------------------------------------------------------------------
extern "C" void kernel_launch(void* const* d_in, const int* in_sizes, int n_in,
                              void* d_out, int out_size, void* d_ws, size_t ws_size,
                              hipStream_t stream) {
    const float* x0   = (const float*)d_in[0];
    const int*   esrc = (const int*)d_in[1];
    const int*   edst = (const int*)d_in[2];
    const float* Wq   = (const float*)d_in[3];
    const float* bq   = (const float*)d_in[4];
    const float* Wk   = (const float*)d_in[5];
    const float* Wv   = (const float*)d_in[6];
    const float* Wo   = (const float*)d_in[7];
    const float* bo   = (const float*)d_in[8];
    const float* g1   = (const float*)d_in[9];
    const float* b1n  = (const float*)d_in[10];
    const float* W1   = (const float*)d_in[11];
    const float* b1   = (const float*)d_in[12];
    const float* W2   = (const float*)d_in[13];
    const float* b2   = (const float*)d_in[14];
    const float* g2   = (const float*)d_in[15];
    const float* b2n  = (const float*)d_in[16];

    const int N = in_sizes[0] / D_MODEL;
    const int E = in_sizes[1];

    // -------- workspace carve-out (256B aligned) --------
    size_t off = 0;
    char* base = (char*)d_ws;
    auto carve = [&](size_t bytes) -> void* {
        void* p = base + off;
        off += (bytes + 255) & ~(size_t)255;
        return p;
    };
    const size_t perLayerW = 4 * 65536 + 2 * 262144;         // bf16 elems/layer
    __bf16* wbf  = (__bf16*)carve((size_t)N_LAYER * perLayerW * sizeof(__bf16));
    __bf16* abf  = (__bf16*)carve((size_t)N * D_MODEL * sizeof(__bf16)); // x/o/h bf16 (reused)
    float*  qb   = (float*) carve((size_t)N * D_MODEL * sizeof(float));
    float*  kb   = (float*) carve((size_t)N * D_MODEL * sizeof(float));
    float*  vb   = (float*) carve((size_t)N * D_MODEL * sizeof(float));
    float*  sc   = (float*) carve((size_t)E * N_HEAD  * sizeof(float));
    float*  wv   = (float*) carve((size_t)N * D_MODEL * sizeof(float));
    float*  zb   = (float*) carve((size_t)N * N_HEAD  * sizeof(float));
    float*  hb   = (float*) carve((size_t)N * D_MODEL * sizeof(float));
    __bf16* ffb  = (__bf16*)carve((size_t)N * D_FF    * sizeof(__bf16));
    float*  xa   = (float*) carve((size_t)N * D_MODEL * sizeof(float));
    float*  xc   = (float*) carve((size_t)N * D_MODEL * sizeof(float));
    float*  tmp1 = qb;   // Wo GEMM output  (q dead after edge_score+aggr)
    float*  tmp2 = kb;   // W2 GEMM output  (k dead after edge_score)

    // -------- one-time weight transpose+convert to bf16 --------
    for (int l = 0; l < N_LAYER; ++l) {
        __bf16* lw = wbf + (size_t)l * perLayerW;
        __bf16* WqT = lw;
        __bf16* WkT = lw + 65536;
        __bf16* WvT = lw + 131072;
        __bf16* WoT = lw + 196608;
        __bf16* W1T = lw + 262144;   // [DFF][D]
        __bf16* W2T = lw + 524288;   // [D][DFF]
        const int g64k  = (65536  + 255) / 256;
        const int g256k = (262144 + 255) / 256;
        transpose_to_bf16_kernel<<<g64k, 256, 0, stream>>>(Wq + (size_t)l * 65536, WqT, D_MODEL, D_MODEL);
        transpose_to_bf16_kernel<<<g64k, 256, 0, stream>>>(Wk + (size_t)l * 65536, WkT, D_MODEL, D_MODEL);
        transpose_to_bf16_kernel<<<g64k, 256, 0, stream>>>(Wv + (size_t)l * 65536, WvT, D_MODEL, D_MODEL);
        transpose_to_bf16_kernel<<<g64k, 256, 0, stream>>>(Wo + (size_t)l * 65536, WoT, D_MODEL, D_MODEL);
        transpose_to_bf16_kernel<<<g256k, 256, 0, stream>>>(W1 + (size_t)l * 262144, W1T, D_MODEL, D_FF);
        transpose_to_bf16_kernel<<<g256k, 256, 0, stream>>>(W2 + (size_t)l * 262144, W2T, D_FF, D_MODEL);
    }

    // initial bf16 activations
    {
        const size_t n = (size_t)N * D_MODEL;
        f32_to_bf16_kernel<<<(int)((n + 255) / 256), 256, 0, stream>>>(x0, abf, n);
    }

    const int gM   = (N + 127) / 128;           // GEMM row blocks
    const dim3 gD  (gM, D_MODEL / 64);          // Nc = 256
    const dim3 gF  (gM, D_FF / 64);             // Nc = 1024
    const int gE   = (E + 7) / 8;               // wave-per-edge grids
    const int gN   = (N + 7) / 8;               // wave-per-node grids

    const float* xcur = x0;
    for (int l = 0; l < N_LAYER; ++l) {
        __bf16* lw  = wbf + (size_t)l * perLayerW;
        __bf16* WqT = lw;
        __bf16* WkT = lw + 65536;
        __bf16* WvT = lw + 131072;
        __bf16* WoT = lw + 196608;
        __bf16* W1T = lw + 262144;
        __bf16* W2T = lw + 524288;

        // Q, K, V projections (bf16 WMMA, f32 out)
        gemm_bf16_wmma_kernel<<<gD, 256, 0, stream>>>(abf, WqT, bq + (size_t)l * D_MODEL,
                                                      qb, nullptr, N, D_MODEL, D_MODEL, 0);
        gemm_bf16_wmma_kernel<<<gD, 256, 0, stream>>>(abf, WkT, nullptr,
                                                      kb, nullptr, N, D_MODEL, D_MODEL, 0);
        gemm_bf16_wmma_kernel<<<gD, 256, 0, stream>>>(abf, WvT, nullptr,
                                                      vb, nullptr, N, D_MODEL, D_MODEL, 0);

        // edge attention
        edge_score_kernel<<<gE, 256, 0, stream>>>(esrc, edst, qb, kb, sc, E);
        {
            const size_t nwv = (size_t)N * D_MODEL;
            const size_t nz  = (size_t)N * N_HEAD;
            zero_f32_kernel<<<(int)((nwv + 255) / 256), 256, 0, stream>>>(wv, nwv);
            zero_f32_kernel<<<(int)((nz  + 255) / 256), 256, 0, stream>>>(zb, nz);
        }
        edge_aggr_kernel<<<gE, 256, 0, stream>>>(esrc, edst, vb, sc, wv, zb, E);
        finalize_o_kernel<<<gN, 256, 0, stream>>>(wv, zb, abf, N);   // abf <- o (bf16)

        // o @ Wo + bo ; h = LN(x + .)   (abf <- h bf16)
        gemm_bf16_wmma_kernel<<<gD, 256, 0, stream>>>(abf, WoT, bo + (size_t)l * D_MODEL,
                                                      tmp1, nullptr, N, D_MODEL, D_MODEL, 0);
        residual_ln_kernel<<<gN, 256, 0, stream>>>(xcur, tmp1,
                                                   g1 + (size_t)l * D_MODEL,
                                                   b1n + (size_t)l * D_MODEL,
                                                   hb, abf, N);

        // FFN: relu(h@W1+b1) -> bf16 ; @W2+b2 ; x' = LN(h + .)
        gemm_bf16_wmma_kernel<<<gF, 256, 0, stream>>>(abf, W1T, b1 + (size_t)l * D_FF,
                                                      nullptr, ffb, N, D_MODEL, D_FF, 1);
        gemm_bf16_wmma_kernel<<<gD, 256, 0, stream>>>(ffb, W2T, b2 + (size_t)l * D_MODEL,
                                                      tmp2, nullptr, N, D_FF, D_MODEL, 0);

        float* xnext = (l == N_LAYER - 1) ? (float*)d_out : ((l & 1) ? xc : xa);
        residual_ln_kernel<<<gN, 256, 0, stream>>>(hb, tmp2,
                                                   g2 + (size_t)l * D_MODEL,
                                                   b2n + (size_t)l * D_MODEL,
                                                   xnext, abf, N);   // abf <- next-layer x (bf16)
        xcur = xnext;
    }
}
